// RelAwareMemory_51788715655449
// MI455X (gfx1250) — matrix operand order repaired and checked by reading, not compiled, not run on hardware
//
#include <hip/hip_runtime.h>
#include <hip/hip_bf16.h>

// ---------------------------------------------------------------------------
// RelAwareMemory layer for MI455X (gfx1250, wave32, WMMA).
// Matmuls: v_wmma_f32_16x16x32_f16. GEMM staging: async global->LDS copies
// (ASYNCcnt) with double-buffered software pipeline. Scores are materialized
// in both q-major (for pq softmax) and k-major (for pv softmax) layouts so
// every score access is a contiguous 16B vector op.
// ---------------------------------------------------------------------------

#define B_   32
#define LQ_  256
#define H_   256
#define NH_  8
#define DH_  32
#define LK_  512
#define N_   8192

typedef _Float16 half_t;
typedef __attribute__((ext_vector_type(16))) _Float16 v16h;
typedef __attribute__((ext_vector_type(8)))  float    v8f;
typedef int v4i __attribute__((vector_size(16)));      // matches builtin param

struct alignas(16) B16 { unsigned long long x, y; };   // POD 16-byte chunk
union FragA  { v16h v; _Float16 h[16]; B16 q[2]; };
union Chunk8 { B16 raw; _Float16 h[8]; };
union Pack2  { _Float16 h[2]; unsigned u; };

#if __has_builtin(__builtin_amdgcn_global_load_async_to_lds_b128)
#define HAVE_ASYNC_LDS 1
#else
#define HAVE_ASYNC_LDS 0
#endif

__device__ __forceinline__ void wait_async0() {
#if __has_builtin(__builtin_amdgcn_s_wait_asynccnt)
  __builtin_amdgcn_s_wait_asynccnt(0);
#else
  asm volatile("s_wait_asynccnt 0x0" ::: "memory");
#endif
}

#if HAVE_ASYNC_LDS
__device__ __forceinline__ void async_cp16(half_t* lds, const half_t* g) {
  __builtin_amdgcn_global_load_async_to_lds_b128(
      (__attribute__((address_space(1))) v4i*)const_cast<half_t*>(g),
      (__attribute__((address_space(3))) v4i*)lds, 0, 0);
}
#endif

__device__ __forceinline__ v8f wmma_f16(const FragA& a, const FragA& b, v8f c) {
  return __builtin_amdgcn_wmma_f32_16x16x32_f16(false, a.v, false, b.v,
                                                (short)0, c, false, false);
}

// ------------------------------- fills -------------------------------------
__global__ void fill_zero16_kernel(B16* p, int n16) {
  int i = blockIdx.x * blockDim.x + threadIdx.x;
  if (i < n16) { B16 z; z.x = 0ULL; z.y = 0ULL; p[i] = z; }
}
__global__ void f32_to_f16_kernel(const float* __restrict__ in, half_t* __restrict__ out, int n) {
  int i = blockIdx.x * blockDim.x + threadIdx.x;
  if (i < n) out[i] = (half_t)in[i];
}

// --------------------------- edge gather / scatter --------------------------
__global__ void edge_gather_kernel(const float* __restrict__ graph, const float* __restrict__ rel,
                                   const int* __restrict__ src, const int* __restrict__ dst,
                                   const int* __restrict__ eb, const int* __restrict__ ep,
                                   half_t* __restrict__ key_pad_h,
                                   half_t* __restrict__ val_pad_h, float* __restrict__ val_pad_f) {
  int e = blockIdx.x;
  int i = threadIdx.x;
  int bb = eb[e], pos = ep[e];
  size_t ro = (size_t)bb * LK_ + pos;
  float gs = graph[(size_t)src[e] * H_ + i];
  key_pad_h[ro * (2 * H_) + i]      = (half_t)gs;
  key_pad_h[ro * (2 * H_) + H_ + i] = (half_t)rel[(size_t)e * H_ + i];
  float gv = graph[(size_t)dst[e] * H_ + i];
  val_pad_f[ro * H_ + i] = gv;
  val_pad_h[ro * H_ + i] = (half_t)gv;
}

__global__ void edge_scatter_add_kernel(const float* __restrict__ value_new,
                                        const int* __restrict__ dst,
                                        const int* __restrict__ eb, const int* __restrict__ ep,
                                        float* __restrict__ sums, float* __restrict__ cnts) {
  int e = blockIdx.x;
  int i = threadIdx.x;
  size_t ro = (size_t)eb[e] * LK_ + ep[e];
  float v = value_new[ro * H_ + i];
  atomicAdd(&sums[(size_t)dst[e] * H_ + i], v);
  if (i == 0) atomicAdd(&cnts[dst[e]], 1.0f);
}

__global__ void finalize_graph_kernel(const float* __restrict__ sums, const float* __restrict__ cnts,
                                      const float* __restrict__ graph, float* __restrict__ out) {
  int nrow = blockIdx.x;
  int i = threadIdx.x;
  float c = cnts[nrow];
  size_t o = (size_t)nrow * H_ + i;
  out[o] = (c > 0.0f) ? sums[o] / c : graph[o];
}

// ------------------------------ WMMA GEMM -----------------------------------
// C[M,Nc] = A(f16)[M,K] @ W(f32)[K,Nc] + bias ; template selects outputs.
// Block tile 128(M) x 64(N), K-stage 64, double-buffered LDS pipeline with
// async global->LDS A staging. 8 waves as 4x2; each wave 32x32 (2x2 wmma).
__device__ __forceinline__ void stage_A(const half_t* __restrict__ A, half_t (*As)[72],
                                        int m0, int K, int kt, int tid) {
  int r  = tid >> 1;
  int cb = (tid & 1) * 32;
  const half_t* ap = A + (size_t)(m0 + r) * K + kt + cb;
  half_t* dp = &As[r][cb];
#if HAVE_ASYNC_LDS
  async_cp16(dp,      ap);
  async_cp16(dp + 8,  ap + 8);
  async_cp16(dp + 16, ap + 16);
  async_cp16(dp + 24, ap + 24);
#else
  const B16* sp = reinterpret_cast<const B16*>(ap);
  B16* d2 = reinterpret_cast<B16*>(dp);
  d2[0] = sp[0]; d2[1] = sp[1]; d2[2] = sp[2]; d2[3] = sp[3];
  if (kt + 64 < K) __builtin_prefetch(ap + 64, 0, 0);  // global_prefetch_b8
#endif
}

__device__ __forceinline__ void stage_W(const float* __restrict__ W, half_t (*Ws)[72],
                                        int n0, int Nc, int kt, int tid) {
  for (int j = 0; j < 8; j++) {
    int p  = tid * 8 + j;        // pair index 0..2047
    int nn = p & 63;
    int k2 = (p >> 6) * 2;       // even k
    Pack2 pk;
    pk.h[0] = (half_t)W[(size_t)(kt + k2)     * Nc + n0 + nn];
    pk.h[1] = (half_t)W[(size_t)(kt + k2 + 1) * Nc + n0 + nn];
    *reinterpret_cast<unsigned*>(&Ws[nn][k2]) = pk.u;
  }
}

template <bool WF, bool WH, bool ACC>
__global__ __launch_bounds__(256) void gemm_wmma_kernel(
    const half_t* __restrict__ A, const float* __restrict__ W, const float* __restrict__ bias,
    float* __restrict__ outF, half_t* __restrict__ outH, int M, int K, int Nc) {
  __shared__ half_t As[2][128][72];   // 72-half (144B) rows keep 16B alignment
  __shared__ half_t Ws[2][64][72];    // transposed: Ws[n][k]

  int tid  = threadIdx.x;
  int lane = tid & 31;
  int w    = tid >> 5;
  int wm   = w >> 1;       // 0..3
  int wn   = w & 1;        // 0..1
  int m0   = blockIdx.y * 128;
  int n0   = blockIdx.x * 64;

  v8f acc[2][2] = {};
  const int base = (lane < 16) ? 0 : 8;    // A half-layout base
  const int kb   = (lane < 16) ? 0 : 16;   // unused: B uses the same base as A

  stage_A(A, As[0], m0, K, 0, tid);
  stage_W(W, Ws[0], n0, Nc, 0, tid);
#if HAVE_ASYNC_LDS
  wait_async0();
#endif
  __syncthreads();

  int buf = 0;
  for (int kt = 0; kt < K; kt += 64) {
    int nkt = kt + 64;
    if (nkt < K) {                      // prefetch next stage into other buffer
      stage_A(A, As[buf ^ 1], m0, K, nkt, tid);
      stage_W(W, Ws[buf ^ 1], n0, Nc, nkt, tid);
    }
    for (int s = 0; s < 2; s++) {
      int ko = s * 32;
      FragA a[2], bf[2];
      for (int i = 0; i < 2; i++) {
        int row = wm * 32 + i * 16 + (lane & 15);
        a[i].q[0] = *reinterpret_cast<const B16*>(&As[buf][row][ko + base]);
        a[i].q[1] = *reinterpret_cast<const B16*>(&As[buf][row][ko + 16 + base]);
      }
      for (int j = 0; j < 2; j++) {
        int col = wn * 32 + j * 16 + (lane & 15);
        bf[j].q[0] = *reinterpret_cast<const B16*>(&Ws[buf][col][ko + base]);
        bf[j].q[1] = *reinterpret_cast<const B16*>(&Ws[buf][col][ko + 16 + base]);
      }
      for (int i = 0; i < 2; i++)
        for (int j = 0; j < 2; j++)
          acc[i][j] = wmma_f16(a[i], bf[j], acc[i][j]);
    }
#if HAVE_ASYNC_LDS
    if (nkt < K) wait_async0();
#endif
    __syncthreads();
    buf ^= 1;
  }

  for (int i = 0; i < 2; i++)
    for (int j = 0; j < 2; j++) {
      int col = n0 + wn * 32 + j * 16 + (lane & 15);
      float bv = bias[col];
      for (int r = 0; r < 8; r++) {
        int row = m0 + wm * 32 + i * 16 + ((lane < 16) ? r : 8 + r);
        size_t o = (size_t)row * Nc + col;
        float v = acc[i][j][r] + bv;
        if (ACC) v += outF[o];
        if (WF)  outF[o] = v;
        if (WH)  outH[o] = (half_t)v;
      }
    }
}

// ------------------------------ scores --------------------------------------
// S [b,h,q,k] and S_T [b,h,k,q] = (Q.K)/sqrt(32) + (1-pad[b,k])*(-1e4).
// Mask fold is exact for both softmaxes (constant along q => shift-invariant).
// Per lane the 8 C-fragment elements are 8 consecutive q at fixed k, so the
// transposed copy costs a single contiguous 16B store per tile.
__global__ __launch_bounds__(256) void scores_wmma_kernel(
    const half_t* __restrict__ Qh, const half_t* __restrict__ Kh,
    const float* __restrict__ pad_mask, half_t* __restrict__ S, half_t* __restrict__ ST) {
  int tid = threadIdx.x;
  int lane = tid & 31;
  int w = tid >> 5;
  int t = blockIdx.x * 8 + w;
  int ktg = t & 7;  t >>= 3;   // group of 4 k-tiles (LK/64)
  int qt  = t & 15; t >>= 4;   // LQ/16
  int h   = t & 7;  t >>= 3;
  int bb  = t;

  int n    = lane & 15;
  int base = (lane < 16) ? 0 : 8;
  int db   = (lane < 16) ? 0 : 16;

  FragA a;
  {
    int q = qt * 16 + n;
    const half_t* p = Qh + ((size_t)(bb * LQ_ + q)) * H_ + h * DH_;
    a.q[0] = *reinterpret_cast<const B16*>(p + base);
    a.q[1] = *reinterpret_cast<const B16*>(p + 16 + base);
  }
  const float sc = 0.17677669529663687f;  // 1/sqrt(32)
  for (int sub = 0; sub < 4; sub++) {
    int kt = ktg * 4 + sub;
    int k  = kt * 16 + n;
    FragA bf;
    const half_t* p = Kh + ((size_t)(bb * LK_ + k)) * H_ + h * DH_;
    bf.q[0] = *reinterpret_cast<const B16*>(p + base);
    bf.q[1] = *reinterpret_cast<const B16*>(p + 16 + base);
    v8f acc = {};
    acc = wmma_f16(a, bf, acc);
    float mv = (1.0f - pad_mask[bb * LK_ + k]) * -10000.0f;

    Chunk8 vals;
    for (int r = 0; r < 8; r++) vals.h[r] = (half_t)(acc[r] * sc + mv);
    // transposed copy: one contiguous 16B store (8 consecutive q, fixed k)
    size_t qbase = (size_t)qt * 16 + ((lane < 16) ? 0 : 8);
    *reinterpret_cast<B16*>(ST + ((size_t)(bb * NH_ + h) * LK_ + k) * LQ_ + qbase) = vals.raw;
    // q-major copy: 8 scalar stores (strided by LK)
    for (int r = 0; r < 8; r++) {
      int qq = qt * 16 + ((lane < 16) ? r : 8 + r);
      S[((size_t)(bb * NH_ + h) * LQ_ + qq) * LK_ + k] = vals.h[r];
    }
  }
}

// --------------------- pq softmax (over k) + P @ V --------------------------
// Each wave owns 32 q-rows (one per lane for the softmax stats); the shared
// V fragment feeds 2 row-tiles (4 wmma per K-chunk). Stats move via __shfl.
__global__ __launch_bounds__(256) void attn_pq_kernel(
    const half_t* __restrict__ S, const half_t* __restrict__ Vh, half_t* __restrict__ qctx) {
  int tid = threadIdx.x;
  int lane = tid & 31;
  int w = tid >> 5;
  int t = blockIdx.x * 8 + w;
  int qg = t & 7;  t >>= 3;    // LQ/32 groups
  int h  = t & 7;  t >>= 3;
  int bb = t;

  const size_t srow0 = ((size_t)(bb * NH_ + h) * LQ_ + qg * 32) * LK_;
  const half_t* own = S + srow0 + (size_t)lane * LK_;   // this lane's row

  float m = -3.0e38f;
  for (int i = 0; i < LK_ / 8; i++) {
    Chunk8 c; c.raw = *reinterpret_cast<const B16*>(own + i * 8);
    for (int j = 0; j < 8; j++) m = fmaxf(m, (float)c.h[j]);
  }
  float s = 0.0f;
  for (int i = 0; i < LK_ / 8; i++) {
    Chunk8 c; c.raw = *reinterpret_cast<const B16*>(own + i * 8);
    for (int j = 0; j < 8; j++) s += __expf((float)c.h[j] - m);
  }

  int base = (lane < 16) ? 0 : 8;
  int kb   = (lane < 16) ? 0 : 16;
  int n = lane & 15;
  const half_t* frow[2];
  float mt[2];
  for (int t2 = 0; t2 < 2; t2++) {
    frow[t2] = S + srow0 + (size_t)(t2 * 16 + (lane & 15)) * LK_;
    mt[t2]   = __shfl(m, t2 * 16 + (lane & 15), 32);
  }
  v8f acc[2][2] = {};
  for (int kc = 0; kc < LK_ / 32; kc++) {
    FragA a[2];
    for (int t2 = 0; t2 < 2; t2++) {
      Chunk8 c0, c1;
      c0.raw = *reinterpret_cast<const B16*>(frow[t2] + kc * 32 + base);
      c1.raw = *reinterpret_cast<const B16*>(frow[t2] + kc * 32 + 16 + base);
      for (int j = 0; j < 8; j++) {
        a[t2].h[j]     = (half_t)__expf((float)c0.h[j] - mt[t2]);
        a[t2].h[8 + j] = (half_t)__expf((float)c1.h[j] - mt[t2]);
      }
    }
    FragA b0, b1;
    const half_t* vp = Vh + ((size_t)(bb * LK_ + kc * 32 + base)) * H_ + h * DH_;
    for (int j = 0; j < 16; j++) {
      int jr = (j < 8) ? j : j + 8;
      b0.h[j] = vp[(size_t)jr * H_ + n];
      b1.h[j] = vp[(size_t)jr * H_ + 16 + n];
    }
    for (int t2 = 0; t2 < 2; t2++) {
      acc[t2][0] = wmma_f16(a[t2], b0, acc[t2][0]);
      acc[t2][1] = wmma_f16(a[t2], b1, acc[t2][1]);
    }
  }
  for (int t2 = 0; t2 < 2; t2++)
    for (int r = 0; r < 8; r++) {
      int rr = (lane < 16) ? r : 8 + r;
      int qq = qg * 32 + t2 * 16 + rr;
      float inv = 1.0f / __shfl(s, t2 * 16 + rr, 32);
      size_t o = ((size_t)(bb * LQ_ + qq)) * H_ + h * DH_;
      qctx[o + n]      = (half_t)(acc[t2][0][r] * inv);
      qctx[o + 16 + n] = (half_t)(acc[t2][1][r] * inv);
    }
}

// --------------------- pv softmax (over q) + P' @ Q -------------------------
// Reads the k-major transposed scores: every access is a contiguous 16B load.
__global__ __launch_bounds__(256) void attn_pv_kernel(
    const half_t* __restrict__ ST, const half_t* __restrict__ Qh,
    const float* __restrict__ emask, half_t* __restrict__ vctx_h, float* __restrict__ vctx_f) {
  int tid = threadIdx.x;
  int lane = tid & 31;
  int w = tid >> 5;
  int t = blockIdx.x * 8 + w;
  int kg = t & 15; t >>= 4;    // LK/32 groups
  int h  = t & 7;  t >>= 3;
  int bb = t;

  const size_t sbase = ((size_t)(bb * NH_ + h) * LK_ + kg * 32) * LQ_;
  const half_t* own = ST + sbase + (size_t)lane * LQ_;  // this lane's k-row
  const float* em = emask + (size_t)bb * LQ_;

  float m = -3.0e38f;
  for (int i = 0; i < LQ_ / 8; i++) {
    Chunk8 c; c.raw = *reinterpret_cast<const B16*>(own + i * 8);
    for (int j = 0; j < 8; j++) m = fmaxf(m, (float)c.h[j] + em[i * 8 + j]);
  }
  float s = 0.0f;
  for (int i = 0; i < LQ_ / 8; i++) {
    Chunk8 c; c.raw = *reinterpret_cast<const B16*>(own + i * 8);
    for (int j = 0; j < 8; j++) s += __expf((float)c.h[j] + em[i * 8 + j] - m);
  }

  int base = (lane < 16) ? 0 : 8;
  int qb   = (lane < 16) ? 0 : 16;
  int n = lane & 15;
  const half_t* frow[2];
  float mt[2];
  for (int t2 = 0; t2 < 2; t2++) {
    frow[t2] = ST + sbase + (size_t)(t2 * 16 + (lane & 15)) * LQ_;
    mt[t2]   = __shfl(m, t2 * 16 + (lane & 15), 32);
  }
  v8f acc[2][2] = {};
  for (int qc = 0; qc < LQ_ / 32; qc++) {
    int q0 = qc * 32 + base;
    int q1 = q0 + 16;
    FragA a[2];
    for (int t2 = 0; t2 < 2; t2++) {
      Chunk8 c0, c1;
      c0.raw = *reinterpret_cast<const B16*>(frow[t2] + q0);
      c1.raw = *reinterpret_cast<const B16*>(frow[t2] + q1);
      for (int j = 0; j < 8; j++) {
        a[t2].h[j]     = (half_t)__expf((float)c0.h[j] + em[q0 + j] - mt[t2]);
        a[t2].h[8 + j] = (half_t)__expf((float)c1.h[j] + em[q1 + j] - mt[t2]);
      }
    }
    FragA b0, b1;
    const half_t* qp = Qh + ((size_t)(bb * LQ_ + qc * 32 + base)) * H_ + h * DH_;
    for (int j = 0; j < 16; j++) {
      int jr = (j < 8) ? j : j + 8;
      b0.h[j] = qp[(size_t)jr * H_ + n];
      b1.h[j] = qp[(size_t)jr * H_ + 16 + n];
    }
    for (int t2 = 0; t2 < 2; t2++) {
      acc[t2][0] = wmma_f16(a[t2], b0, acc[t2][0]);
      acc[t2][1] = wmma_f16(a[t2], b1, acc[t2][1]);
    }
  }
  for (int t2 = 0; t2 < 2; t2++)
    for (int r = 0; r < 8; r++) {
      int rr = (lane < 16) ? r : 8 + r;
      int k2 = kg * 32 + t2 * 16 + rr;
      float inv = 1.0f / __shfl(s, t2 * 16 + rr, 32);
      size_t o = ((size_t)(bb * LK_ + k2)) * H_ + h * DH_;
      float v0 = acc[t2][0][r] * inv, v1 = acc[t2][1][r] * inv;
      vctx_f[o + n] = v0;          vctx_f[o + 16 + n] = v1;
      vctx_h[o + n] = (half_t)v0;  vctx_h[o + 16 + n] = (half_t)v1;
    }
}

// ------------------------------ layernorms ----------------------------------
__global__ __launch_bounds__(256) void ln_residual_kernel(
    const float* __restrict__ x_in, const float* __restrict__ resid,
    const float* __restrict__ g, const float* __restrict__ beta, float* __restrict__ out) {
  int row = blockIdx.x;
  int i = threadIdx.x;
  size_t o = (size_t)row * H_ + i;
  float x = x_in[o] + resid[o];
  __shared__ float red[256];
  red[i] = x; __syncthreads();
  for (int st = 128; st > 0; st >>= 1) { if (i < st) red[i] += red[i + st]; __syncthreads(); }
  float mean = red[0] * (1.0f / H_);
  __syncthreads();
  float d = x - mean;
  red[i] = d * d; __syncthreads();
  for (int st = 128; st > 0; st >>= 1) { if (i < st) red[i] += red[i + st]; __syncthreads(); }
  float var = red[0] * (1.0f / H_);
  out[o] = d * rsqrtf(var + 1e-12f) * g[i] + beta[i];
}

__global__ __launch_bounds__(256) void gate_ln_kernel(
    const float* __restrict__ tg, const float* __restrict__ vctx_f, const float* __restrict__ val_pad_f,
    const float* __restrict__ g, const float* __restrict__ beta, float* __restrict__ out) {
  int row = blockIdx.x;
  int i = threadIdx.x;
  size_t o = (size_t)row * H_ + i;
  float th = 1.0f / (1.0f + __expf(-tg[o]));
  float x = th * vctx_f[o] + (1.0f - th) * val_pad_f[o];
  __shared__ float red[256];
  red[i] = x; __syncthreads();
  for (int st = 128; st > 0; st >>= 1) { if (i < st) red[i] += red[i + st]; __syncthreads(); }
  float mean = red[0] * (1.0f / H_);
  __syncthreads();
  float d = x - mean;
  red[i] = d * d; __syncthreads();
  for (int st = 128; st > 0; st >>= 1) { if (i < st) red[i] += red[i + st]; __syncthreads(); }
  float var = red[0] * (1.0f / H_);
  out[o] = d * rsqrtf(var + 1e-12f) * g[i] + beta[i];
}

// ------------------------------- launcher -----------------------------------
extern "C" void kernel_launch(void* const* d_in, const int* in_sizes, int n_in,
                              void* d_out, int out_size, void* d_ws, size_t ws_size,
                              hipStream_t stream) {
  (void)n_in; (void)out_size; (void)ws_size;
  const float* emask = (const float*)d_in[0];
  const float* query = (const float*)d_in[1];
  const float* rel   = (const float*)d_in[2];
  const float* graph = (const float*)d_in[3];
  const int*   src   = (const int*)d_in[4];
  const int*   dst   = (const int*)d_in[5];
  const int*   eb    = (const int*)d_in[6];
  const int*   ep    = (const int*)d_in[7];
  const float* pad   = (const float*)d_in[8];
  const float* Wq = (const float*)d_in[9];   const float* bq = (const float*)d_in[10];
  const float* Wk = (const float*)d_in[11];  const float* bk = (const float*)d_in[12];
  const float* Wv = (const float*)d_in[13];  const float* bv = (const float*)d_in[14];
  const float* Wo = (const float*)d_in[15];  const float* bo = (const float*)d_in[16];
  const float* ln1g = (const float*)d_in[17]; const float* ln1b = (const float*)d_in[18];
  const float* W1 = (const float*)d_in[19];  const float* b1 = (const float*)d_in[20];
  const float* W2 = (const float*)d_in[21];  const float* b2 = (const float*)d_in[22];
  const float* ln2g = (const float*)d_in[23]; const float* ln2b = (const float*)d_in[24];
  int E = in_sizes[4];

  float* out_graph = (float*)d_out;
  float* out_query = (float*)d_out + (size_t)N_ * H_;

  char* ws = (char*)d_ws;
  size_t off = 0;
  auto take = [&](size_t bytes) -> char* {
    char* p = ws + off;
    off += (bytes + 255) & ~(size_t)255;
    return p;
  };
  const size_t BLK = (size_t)B_ * LK_, BLQ = (size_t)B_ * LQ_;
  half_t* key_pad_h = (half_t*)take(BLK * 2 * H_ * 2);
  half_t* val_pad_h = (half_t*)take(BLK * H_ * 2);
  float*  val_pad_f = (float*)take(BLK * H_ * 4);
  half_t* query_h   = (half_t*)take(BLQ * H_ * 2);
  half_t* Qh        = (half_t*)take(BLQ * H_ * 2);
  half_t* Kh        = (half_t*)take(BLK * H_ * 2);
  half_t* Vh        = (half_t*)take(BLK * H_ * 2);
  half_t* S         = (half_t*)take((size_t)B_ * NH_ * LQ_ * LK_ * 2);
  half_t* ST        = (half_t*)take((size_t)B_ * NH_ * LQ_ * LK_ * 2);
  half_t* qctx_h    = (half_t*)take(BLQ * H_ * 2);
  float*  wo_out    = (float*)take(BLQ * H_ * 4);
  half_t* vctx_h    = (half_t*)take(BLK * H_ * 2);
  float*  vctx_f    = (float*)take(BLK * H_ * 4);
  float*  tgate     = (float*)take(BLK * H_ * 4);
  float*  value_new = (float*)take(BLK * H_ * 4);
  float*  sums      = (float*)take((size_t)N_ * H_ * 4);
  float*  cnts      = (float*)take((size_t)N_ * 4);

  // 1) zero padded buffers (16B-vector fills), gather edges, query -> f16
  {
    int n16 = (int)(BLK * 2 * H_ * 2 / 16);
    fill_zero16_kernel<<<(n16 + 255) / 256, 256, 0, stream>>>((B16*)key_pad_h, n16);
    n16 = (int)(BLK * H_ * 2 / 16);
    fill_zero16_kernel<<<(n16 + 255) / 256, 256, 0, stream>>>((B16*)val_pad_h, n16);
    n16 = (int)(BLK * H_ * 4 / 16);
    fill_zero16_kernel<<<(n16 + 255) / 256, 256, 0, stream>>>((B16*)val_pad_f, n16);
  }
  edge_gather_kernel<<<E, 256, 0, stream>>>(graph, rel, src, dst, eb, ep,
                                            key_pad_h, val_pad_h, val_pad_f);
  {
    int n = (int)(BLQ * H_);
    f32_to_f16_kernel<<<(n + 255) / 256, 256, 0, stream>>>(query, query_h, n);
  }

  // 2) projections (WMMA GEMMs, async-LDS pipelined)
  dim3 gq(H_ / 64, (unsigned)(BLQ / 128));
  dim3 gk(H_ / 64, (unsigned)(BLK / 128));
  gemm_wmma_kernel<false, true, false><<<gq, 256, 0, stream>>>(query_h,   Wq, bq, nullptr, Qh, (int)BLQ, H_,     H_);
  gemm_wmma_kernel<false, true, false><<<gk, 256, 0, stream>>>(key_pad_h, Wk, bk, nullptr, Kh, (int)BLK, 2 * H_, H_);
  gemm_wmma_kernel<false, true, false><<<gk, 256, 0, stream>>>(val_pad_h, Wv, bv, nullptr, Vh, (int)BLK, H_,     H_);

  // 3) scores with folded pad mask, dual layout (one wmma per 16x16 tile)
  {
    int tiles = B_ * NH_ * (LQ_ / 16) * (LK_ / 64);  // waves; 4 k-tiles each
    scores_wmma_kernel<<<tiles / 8, 256, 0, stream>>>(Qh, Kh, pad, S, ST);
  }

  // 4) two softmax+context passes
  attn_pq_kernel<<<B_ * NH_ * (LQ_ / 32) / 8, 256, 0, stream>>>(S, Vh, qctx_h);
  attn_pv_kernel<<<B_ * NH_ * (LK_ / 32) / 8, 256, 0, stream>>>(ST, Qh, emask, vctx_h, vctx_f);

  // 5) output projection + residual layernorm -> query_out
  gemm_wmma_kernel<true, false, false><<<gq, 256, 0, stream>>>(qctx_h, Wo, bo, wo_out, nullptr, (int)BLQ, H_, H_);
  ln_residual_kernel<<<(unsigned)BLQ, 256, 0, stream>>>(wo_out, query, ln1g, ln1b, out_query);

  // 6) gate: t = vctx@W1 + b1 + val_pad@W2 + b2 ; value_new = LN(theta*vctx + (1-theta)*val_pad)
  gemm_wmma_kernel<true, false, false><<<gk, 256, 0, stream>>>(vctx_h,    W1, b1, tgate, nullptr, (int)BLK, H_, H_);
  gemm_wmma_kernel<true, false, true ><<<gk, 256, 0, stream>>>(val_pad_h, W2, b2, tgate, nullptr, (int)BLK, H_, H_);
  gate_ln_kernel<<<(unsigned)BLK, 256, 0, stream>>>(tgate, vctx_f, val_pad_f, ln2g, ln2b, value_new);

  // 7) scatter-mean by dst -> graph_new
  {
    int n16 = (int)((size_t)N_ * H_ * 4 / 16);
    fill_zero16_kernel<<<(n16 + 255) / 256, 256, 0, stream>>>((B16*)sums, n16);
    n16 = (int)((size_t)N_ * 4 / 16);
    fill_zero16_kernel<<<(n16 + 255) / 256, 256, 0, stream>>>((B16*)cnts, n16);
  }
  edge_scatter_add_kernel<<<E, 256, 0, stream>>>(value_new, dst, eb, ep, sums, cnts);
  finalize_graph_kernel<<<N_, 256, 0, stream>>>(sums, cnts, graph, out_graph);
}
